// Attention_28484223108010
// MI455X (gfx1250) — matrix operand order repaired and checked by reading, not compiled, not run on hardware
//
#include <hip/hip_runtime.h>
#include <hip/hip_bf16.h>

// MI455X / gfx1250 attention block.
// Pipeline: LN+cast -> weight cast/transpose -> fused QKV WMMA GEMM (16x64/wave)
//           -> flash attention (async double-buffered K/V LDS staging,
//              WMMA QK^T + online softmax + WMMA PV) -> out-proj WMMA GEMM.
// All matmuls use v_wmma_f32_16x16x32_f16 (wave32).

typedef __attribute__((ext_vector_type(16))) _Float16 v16h;
typedef __attribute__((ext_vector_type(8)))  _Float16 v8h;
typedef __attribute__((ext_vector_type(8)))  float    v8f;
typedef int v4i __attribute__((vector_size(16)));   // matches builtin prototype

#define BATCH 2
#define SEQ   2048
#define DIM   1024
#define HEADS 16
#define DHEAD 64
#define TOK   (BATCH * SEQ)      // 4096
#define NQKV  (3 * DIM)          // 3072 (q|k|v columns)
#define SCALE 0.125f             // 64^-0.5

// ---- async global->LDS copy (ASYNCcnt path), with synchronous fallback ----
#if __has_builtin(__builtin_amdgcn_global_load_async_to_lds_b128) && \
    __has_builtin(__builtin_amdgcn_s_wait_asynccnt)
#define HAVE_ASYNC_LDS 1
#else
#define HAVE_ASYNC_LDS 0
#endif

#if HAVE_ASYNC_LDS
__device__ __forceinline__ void copy16(const _Float16* g, _Float16* l) {
    __builtin_amdgcn_global_load_async_to_lds_b128(
        (__attribute__((address_space(1))) v4i*)g,
        (__attribute__((address_space(3))) v4i*)l, 0, 0);
}
__device__ __forceinline__ void wait_copies_tail() {  // current tile done, next in flight
    __builtin_amdgcn_s_wait_asynccnt(2);
}
__device__ __forceinline__ void wait_copies_all() {
    __builtin_amdgcn_s_wait_asynccnt(0);
}
#else
__device__ __forceinline__ void copy16(const _Float16* g, _Float16* l) {
    *(v8h*)l = *(const v8h*)g;
}
__device__ __forceinline__ void wait_copies_tail() {}
__device__ __forceinline__ void wait_copies_all() {}
#endif

// Build a v16h A/B fragment from two contiguous 16B chunks.
__device__ __forceinline__ v16h mk16(const _Float16* lo, const _Float16* hi) {
    v8h a = *(const v8h*)lo;
    v8h b = *(const v8h*)hi;
    v16h r;
#pragma unroll
    for (int e = 0; e < 8; ++e) { r[e] = a[e]; r[8 + e] = b[e]; }
    return r;
}

__device__ __forceinline__ float rowmax16(float v) {
    v = fmaxf(v, __shfl_xor(v, 1, 16));
    v = fmaxf(v, __shfl_xor(v, 2, 16));
    v = fmaxf(v, __shfl_xor(v, 4, 16));
    v = fmaxf(v, __shfl_xor(v, 8, 16));
    return v;
}
__device__ __forceinline__ float rowsum16(float v) {
    v += __shfl_xor(v, 1, 16);
    v += __shfl_xor(v, 2, 16);
    v += __shfl_xor(v, 4, 16);
    v += __shfl_xor(v, 8, 16);
    return v;
}

// ---------------- Kernel 1: LayerNorm + cast to f16 ----------------
__global__ void __launch_bounds__(256) ln_kernel(const float* __restrict__ x,
                                                 const float* __restrict__ gamma,
                                                 const float* __restrict__ beta,
                                                 _Float16* __restrict__ xn) {
    __shared__ float red[256];
    __shared__ float red2[256];
    const int t = blockIdx.x;
    const int tid = threadIdx.x;
    const float4 v = *(const float4*)(x + (size_t)t * DIM + tid * 4);
    float s  = v.x + v.y + v.z + v.w;
    float sq = v.x * v.x + v.y * v.y + v.z * v.z + v.w * v.w;
    red[tid] = s; red2[tid] = sq;
    __syncthreads();
#pragma unroll
    for (int off = 128; off > 0; off >>= 1) {
        if (tid < off) { red[tid] += red[tid + off]; red2[tid] += red2[tid + off]; }
        __syncthreads();
    }
    const float mu   = red[0] * (1.0f / DIM);
    const float var  = red2[0] * (1.0f / DIM) - mu * mu;
    const float rstd = rsqrtf(var + 1e-5f);
    const float vv[4] = {v.x, v.y, v.z, v.w};
#pragma unroll
    for (int e = 0; e < 4; ++e) {
        const int c = tid * 4 + e;
        xn[(size_t)t * DIM + c] = (_Float16)((vv[e] - mu) * rstd * gamma[c] + beta[c]);
    }
}

// ------------- Kernel 2: cast weights to f16, transposed [N,K] -------------
__global__ void __launch_bounds__(256) castw_kernel(const float* __restrict__ wqk,
                                                    const float* __restrict__ wv,
                                                    const float* __restrict__ wo,
                                                    _Float16* __restrict__ wqkvT,
                                                    _Float16* __restrict__ woT) {
    const int idx = blockIdx.x * 256 + threadIdx.x;   // 0 .. 4M-1
    if (idx < NQKV * DIM) {                           // 3072*1024
        const int c = idx >> 10, kk = idx & 1023;
        const float v = (c < 2 * DIM) ? wqk[(size_t)kk * (2 * DIM) + c]
                                      : wv[(size_t)kk * DIM + (c - 2 * DIM)];
        wqkvT[idx] = (_Float16)v;
    } else {
        const int j = idx - NQKV * DIM;               // 0 .. 1M-1
        const int c = j >> 10, kk = j & 1023;
        woT[j] = (_Float16)wo[(size_t)kk * DIM + c];
    }
}

// ------------- Kernel 3: fused QKV GEMM (xn[4096,1024] @ wT) -------------
// 16x64 output tile per wave (A-fragment reused across 4 WMMAs). The 64-wide
// tile lies inside exactly one head of exactly one of {q,k,v}: branch hoisted.
__global__ void __launch_bounds__(256) gemm_qkv(const _Float16* __restrict__ xn,
                                                const _Float16* __restrict__ wT,
                                                _Float16* __restrict__ q,
                                                _Float16* __restrict__ k,
                                                _Float16* __restrict__ vt) {
    const int lane = threadIdx.x & 31, wave = threadIdx.x >> 5;
    const int tile = blockIdx.x * 8 + wave;           // 12288 tiles total
    const int NT = NQKV / 64;                         // 48
    const int mt = tile / NT, nt = tile % NT;
    const int row0 = mt * 16, col0 = nt * 64;
    const int lh = lane & 15, hi = lane >> 4;
    const _Float16* aBase = xn + (size_t)(row0 + lh) * DIM;
    const _Float16* bBase = wT + (size_t)(col0 + lh) * DIM;
    v8f acc[4] = {};
#pragma unroll 4
    for (int k0 = 0; k0 < DIM; k0 += 32) {
        __builtin_prefetch(aBase + k0 + 256, 0, 1);
        v16h a = mk16(aBase + k0 + hi * 8, aBase + k0 + hi * 8 + 16);
#pragma unroll
        for (int g = 0; g < 4; ++g) {
            const _Float16* bp = bBase + (size_t)g * 16 * DIM + k0 + hi * 8;
            v16h b = mk16(bp, bp + 16);
            acc[g] = __builtin_amdgcn_wmma_f32_16x16x32_f16(false, a, false, b,
                                                            (short)0, acc[g], false, false);
        }
    }
    const int region = col0 >> 10;                    // 0=q, 1=k, 2=v
    const int head = (col0 & 1023) >> 6;
    if (region == 0) {
#pragma unroll
        for (int g = 0; g < 4; ++g) {
            const int dc = g * 16 + lh;
#pragma unroll
            for (int r = 0; r < 8; ++r) {
                const int row = row0 + r + hi * 8;
                const int bi = row >> 11, ni = row & 2047;
                q[((size_t)(bi * HEADS + head) * SEQ + ni) * DHEAD + dc] =
                    (_Float16)(acc[g][r] * SCALE);
            }
        }
    } else if (region == 1) {
#pragma unroll
        for (int g = 0; g < 4; ++g) {
            const int dc = g * 16 + lh;
#pragma unroll
            for (int r = 0; r < 8; ++r) {
                const int row = row0 + r + hi * 8;
                const int bi = row >> 11, ni = row & 2047;
                k[((size_t)(bi * HEADS + head) * SEQ + ni) * DHEAD + dc] =
                    (_Float16)acc[g][r];
            }
        }
    } else {                                          // V stored transposed [d,n]
#pragma unroll
        for (int g = 0; g < 4; ++g) {
            const int dc = g * 16 + lh;
#pragma unroll
            for (int r = 0; r < 8; ++r) {
                const int row = row0 + r + hi * 8;
                const int bi = row >> 11, ni = row & 2047;
                vt[((size_t)(bi * HEADS + head) * DHEAD + dc) * SEQ + ni] =
                    (_Float16)acc[g][r];
            }
        }
    }
}

// ------------- Kernel 4: flash attention -------------
// Block = one (b,h) x 128 query rows (8 waves x 16 rows). K/V tiles for each
// 32-key step are staged in LDS once per block via async copies, double-
// buffered so copy of tile j+1 overlaps WMMAs of tile j.
__global__ void __launch_bounds__(256) attn_kernel(const _Float16* __restrict__ q,
                                                   const _Float16* __restrict__ k,
                                                   const _Float16* __restrict__ vt,
                                                   _Float16* __restrict__ attn) {
    __shared__ __align__(16) _Float16 kbuf[2][32][72];  // keys x d (padded rows)
    __shared__ __align__(16) _Float16 vbuf[2][64][40];  // d x keys (padded rows)
    __shared__ __align__(16) _Float16 pbuf[8][16][40];  // per-wave P staging
    const int tid = threadIdx.x;
    const int lane = tid & 31, wave = tid >> 5;
    const int lh = lane & 15, hi = lane >> 4;
    const int bh = blockIdx.x >> 4;                     // 0..31 (b*h)
    const int i0 = (blockIdx.x & 15) * 128 + wave * 16; // query row tile
    const _Float16* qh = q  + (size_t)bh * SEQ * DHEAD;
    const _Float16* kh = k  + (size_t)bh * SEQ * DHEAD;
    const _Float16* vh = vt + (size_t)bh * DHEAD * SEQ;

    // Cooperative copy mapping: 256 threads x 16B covers each 4KB tile.
    const int krow = tid >> 3, kchunk = (tid & 7) * 8;  // 32 rows x 64 halves
    const int vrow = tid >> 2, vchunk = (tid & 3) * 8;  // 64 rows x 32 halves

    // Prologue: start copy of tile 0.
    copy16(kh + (size_t)krow * DHEAD + kchunk, &kbuf[0][krow][kchunk]);
    copy16(vh + (size_t)vrow * SEQ + vchunk,   &vbuf[0][vrow][vchunk]);

    // Q A-fragments for both 32-wide K-steps over d=64 (loaded once).
    v16h aq[2];
#pragma unroll
    for (int ks = 0; ks < 2; ++ks) {
        const _Float16* p = qh + (size_t)(i0 + lh) * DHEAD + ks * 32 + hi * 8;
        aq[ks] = mk16(p, p + 16);
    }

    v8f o[4] = {};                 // 16x64 f32 accumulator (4 N-groups)
    float m[8], l[8];
#pragma unroll
    for (int r = 0; r < 8; ++r) { m[r] = -1e30f; l[r] = 0.0f; }

    for (int jt = 0; jt < SEQ / 32; ++jt) {
        const int cur = jt & 1;
        const int j0 = jt * 32;
        if (jt + 1 < SEQ / 32) {   // launch next tile's copy, then wait for cur
            const int nj = j0 + 32;
            copy16(kh + (size_t)(nj + krow) * DHEAD + kchunk, &kbuf[cur ^ 1][krow][kchunk]);
            copy16(vh + (size_t)vrow * SEQ + nj + vchunk,     &vbuf[cur ^ 1][vrow][vchunk]);
            wait_copies_tail();
        } else {
            wait_copies_all();
        }
        __syncthreads();           // staged K/V visible to all waves

        // S = Q K^T for the 32 keys (B-fragments from LDS).
        v8f s0 = {}, s1 = {};
#pragma unroll
        for (int ks = 0; ks < 2; ++ks) {
            v16h b0 = mk16(&kbuf[cur][lh][ks * 32 + hi * 8],
                           &kbuf[cur][lh][ks * 32 + hi * 8 + 16]);
            v16h b1 = mk16(&kbuf[cur][16 + lh][ks * 32 + hi * 8],
                           &kbuf[cur][16 + lh][ks * 32 + hi * 8 + 16]);
            s0 = __builtin_amdgcn_wmma_f32_16x16x32_f16(false, aq[ks], false, b0,
                                                        (short)0, s0, false, false);
            s1 = __builtin_amdgcn_wmma_f32_16x16x32_f16(false, aq[ks], false, b1,
                                                        (short)0, s1, false, false);
        }
        // Online softmax across the 32 new columns (rows live on 16 lanes).
#pragma unroll
        for (int r = 0; r < 8; ++r) {
            float tmax = rowmax16(fmaxf(s0[r], s1[r]));
            const float mn = fmaxf(m[r], tmax);
            const float corr = __expf(m[r] - mn);
            const float p0 = __expf(s0[r] - mn);
            const float p1 = __expf(s1[r] - mn);
            l[r] = l[r] * corr + rowsum16(p0 + p1);
            m[r] = mn;
#pragma unroll
            for (int g = 0; g < 4; ++g) o[g][r] *= corr;
            const int M = r + hi * 8;
            pbuf[wave][M][lh]      = (_Float16)p0;
            pbuf[wave][M][16 + lh] = (_Float16)p1;
        }
        __syncthreads();
        // Re-read P in A-fragment layout (C-layout -> A-layout via LDS).
        const _Float16* pp = &pbuf[wave][lh][hi * 8];
        v16h ap = mk16(pp, pp + 16);
        // O += P V  (contraction over the 32 keys; 4 N-groups of d)
#pragma unroll
        for (int g = 0; g < 4; ++g) {
            v16h bv = mk16(&vbuf[cur][g * 16 + lh][hi * 8],
                           &vbuf[cur][g * 16 + lh][hi * 8 + 16]);
            o[g] = __builtin_amdgcn_wmma_f32_16x16x32_f16(false, ap, false, bv,
                                                          (short)0, o[g], false, false);
        }
        __syncthreads();           // protect kbuf/vbuf/pbuf before next overwrite
    }
    // Epilogue: normalize and write [token, h*d] f16 for the out-projection.
    const int bi = bh >> 4, head = bh & 15;
#pragma unroll
    for (int g = 0; g < 4; ++g) {
#pragma unroll
        for (int r = 0; r < 8; ++r) {
            const int row = i0 + r + hi * 8;
            const int t = bi * SEQ + row;
            const int col = head * DHEAD + g * 16 + lh;
            attn[(size_t)t * DIM + col] = (_Float16)(o[g][r] / l[r]);
        }
    }
}

// ------------- Kernel 5: out projection + bias (f32 out), 16x64/wave -------------
__global__ void __launch_bounds__(256) gemm_out(const _Float16* __restrict__ attn,
                                                const _Float16* __restrict__ woT,
                                                const float* __restrict__ bout,
                                                float* __restrict__ out) {
    const int lane = threadIdx.x & 31, wave = threadIdx.x >> 5;
    const int tile = blockIdx.x * 8 + wave;           // 4096 tiles
    const int NT = DIM / 64;                          // 16
    const int mt = tile / NT, nt = tile % NT;
    const int row0 = mt * 16, col0 = nt * 64;
    const int lh = lane & 15, hi = lane >> 4;
    const _Float16* aBase = attn + (size_t)(row0 + lh) * DIM;
    const _Float16* bBase = woT + (size_t)(col0 + lh) * DIM;
    v8f acc[4] = {};
#pragma unroll 4
    for (int k0 = 0; k0 < DIM; k0 += 32) {
        __builtin_prefetch(aBase + k0 + 256, 0, 1);
        v16h a = mk16(aBase + k0 + hi * 8, aBase + k0 + hi * 8 + 16);
#pragma unroll
        for (int g = 0; g < 4; ++g) {
            const _Float16* bp = bBase + (size_t)g * 16 * DIM + k0 + hi * 8;
            v16h b = mk16(bp, bp + 16);
            acc[g] = __builtin_amdgcn_wmma_f32_16x16x32_f16(false, a, false, b,
                                                            (short)0, acc[g], false, false);
        }
    }
#pragma unroll
    for (int g = 0; g < 4; ++g) {
        const int col = col0 + g * 16 + lh;
        const float bias = bout[col];
#pragma unroll
        for (int r = 0; r < 8; ++r) {
            const int row = row0 + r + hi * 8;
            out[(size_t)row * DIM + col] = acc[g][r] + bias;
        }
    }
}

extern "C" void kernel_launch(void* const* d_in, const int* in_sizes, int n_in,
                              void* d_out, int out_size, void* d_ws, size_t ws_size,
                              hipStream_t stream) {
    const float* x     = (const float*)d_in[0];   // [2,2048,1024]
    const float* gamma = (const float*)d_in[1];   // [1024]
    const float* beta  = (const float*)d_in[2];   // [1024]
    const float* wqk   = (const float*)d_in[3];   // [1024,2048]
    const float* wv    = (const float*)d_in[4];   // [1024,1024]
    const float* wo    = (const float*)d_in[5];   // [1024,1024]
    const float* bo    = (const float*)d_in[6];   // [1024]
    float* out = (float*)d_out;

    // Workspace layout (48 MiB total, fully rewritten every call):
    char* ws = (char*)d_ws;
    _Float16* xn    = (_Float16*)(ws);                      //  8 MiB  [4096,1024]
    _Float16* wqkvT = (_Float16*)(ws + (size_t)( 8 << 20)); //  6 MiB  [3072,1024]
    _Float16* woT   = (_Float16*)(ws + (size_t)(14 << 20)); //  2 MiB  [1024,1024]
    _Float16* q     = (_Float16*)(ws + (size_t)(16 << 20)); //  8 MiB  [b,h,n,d]
    _Float16* k     = (_Float16*)(ws + (size_t)(24 << 20)); //  8 MiB  [b,h,n,d]
    _Float16* vt    = (_Float16*)(ws + (size_t)(32 << 20)); //  8 MiB  [b,h,d,n]
    _Float16* attn  = (_Float16*)(ws + (size_t)(40 << 20)); //  8 MiB  [4096,1024]

    ln_kernel   <<<TOK,   256, 0, stream>>>(x, gamma, beta, xn);
    castw_kernel<<<16384, 256, 0, stream>>>(wqk, wv, wo, wqkvT, woT);
    gemm_qkv    <<<1536,  256, 0, stream>>>(xn, wqkvT, q, k, vt);   // 12288/8 tiles
    attn_kernel <<<512,   256, 0, stream>>>(q, k, vt, attn);
    gemm_out    <<<512,   256, 0, stream>>>(attn, woT, bo, out);    // 4096/8 tiles
}
